// ST_GraphAttn_26525718020113
// MI455X (gfx1250) — compile-verified, matrix operand order checked by reading, not run on hardware
//
#include <hip/hip_runtime.h>
#include <hip/hip_bf16.h>
#include <math.h>

// ---------------------------------------------------------------------------
// Problem constants (from reference)
// ---------------------------------------------------------------------------
#define N_NODES 8192
#define T_STEPS 64
#define NEDGE   262144
#define H1C     256
#define H2C     256
#define H3C     256
#define H4C     128
#define ATT_HC  16

typedef __attribute__((ext_vector_type(16))) __bf16 v16bf;
typedef __attribute__((ext_vector_type(8)))  float  v8f;

union Frag16 { v16bf v; uint4 q[2]; };

__device__ __forceinline__ __bf16 f2bf(float f) {
  union { float f; unsigned u; } a; a.f = f;
  unsigned r = a.u + 0x7FFFu + ((a.u >> 16) & 1u);   // round-to-nearest-even
  unsigned short h = (unsigned short)(r >> 16);
  __bf16 b; __builtin_memcpy(&b, &h, 2); return b;
}
__device__ __forceinline__ float bf2f(__bf16 b) {
  unsigned short h; __builtin_memcpy(&h, &b, 2);
  union { unsigned u; float f; } a; a.u = ((unsigned)h) << 16; return a.f;
}
// fast reciprocal: single v_rcp_f32 (~1ulp) -- results quantize to bf16 anyway
__device__ __forceinline__ float rcp_(float x) { return __builtin_amdgcn_rcpf(x); }
__device__ __forceinline__ float sigmoidf_(float x) { return rcp_(1.0f + __expf(-x)); }
// branch-free tanh: avoids libm's EXEC-masked two-path expansion in hot loops
__device__ __forceinline__ float tanhf_(float x) {
  float t = __expf(-2.0f * fabsf(x));        // exact 0 at |x| large -> r = 1
  float r = (1.0f - t) * rcp_(1.0f + t);
  return copysignf(r, x);
}

// ---------------------------------------------------------------------------
// Conversions
// ---------------------------------------------------------------------------
__global__ void k_f32_to_bf16(__bf16* __restrict__ dst, const float* __restrict__ src,
                              long long n) {
  long long i = ((long long)blockIdx.x * blockDim.x + threadIdx.x) * 4;
  if (i + 3 < n) {
    float4 f = *(const float4*)(src + i);
    union { __bf16 b[4]; uint2 u; } o;
    o.b[0] = f2bf(f.x); o.b[1] = f2bf(f.y); o.b[2] = f2bf(f.z); o.b[3] = f2bf(f.w);
    *(uint2*)(dst + i) = o.u;
  } else {
    for (long long j = i; j < n; ++j) dst[j] = f2bf(src[j]);
  }
}

// dst (C x R) = transpose of src (R x C), bf16
__global__ void k_f32_to_bf16_T(__bf16* __restrict__ dst, const float* __restrict__ src,
                                int R, int C) {
  int i = blockIdx.x * blockDim.x + threadIdx.x;
  if (i < R * C) {
    int r = i / C, c = i % C;
    dst[(size_t)c * R + r] = f2bf(src[i]);
  }
}

// t1T[j][n] = xc[n] * W1[j]  (t1 = xc @ W1, stored transposed as B^T for WMMA)
__global__ void k_build_t1T(__bf16* __restrict__ t1T, const float* __restrict__ x,
                            const float* __restrict__ W1) {
  int i = blockIdx.x * blockDim.x + threadIdx.x;
  if (i < H1C * N_NODES) {
    int j = i / N_NODES, n = i % N_NODES;
    t1T[i] = f2bf(x[(size_t)n * T_STEPS + (T_STEPS - 1)] * W1[j]);
  }
}

// ---------------------------------------------------------------------------
// Persistent LSTM kernel: 32 rows/block, 8 waves (2/SIMD -> large VGPR budget),
// h state double-buffered in LDS (bf16), c state in registers, recurrent GEMM
// h @ W_hh^T via v_wmma_f32_16x16x32_bf16. Wave w owns M-tile (w>>2) and
// column groups cg = (w&3)*4 + p, p in [0,4). One barrier per timestep.
// ---------------------------------------------------------------------------
__global__ void __launch_bounds__(256)
k_lstm(const float* __restrict__ x, const __bf16* __restrict__ Whh,
       const float* __restrict__ Wih, const float* __restrict__ bih,
       const float* __restrict__ bhh, float* __restrict__ hout) {
  __shared__ __bf16 hbuf[2][32 * 256];
  const int tid = threadIdx.x;
  const int wave = tid >> 5, lane = tid & 31;
  const int l15 = lane & 15, hi = lane >> 4;
  const int rowbase = blockIdx.x * 32;
  const int mtile = wave >> 2;          // waves 0-3 -> rows 0-15, waves 4-7 -> 16-31
  const int cgbase = (wave & 3) * 4;    // pair p -> column group cgbase+p

  for (int i = tid; i < 32 * 256; i += 256) hbuf[0][i] = f2bf(0.0f);

  v8f zero = {};
  v8f cst[4];
#pragma unroll
  for (int p = 0; p < 4; ++p) cst[p] = zero;

  // time-invariant scalars: x-weight and bias per (pair, gate)
  float wij[4][4], bs[4][4];
#pragma unroll
  for (int p = 0; p < 4; ++p) {
    const int cg = cgbase + p;
#pragma unroll
    for (int q = 0; q < 4; ++q) {
      const int j = q * 256 + cg * 16 + l15;
      wij[p][q] = Wih[j];
      bs[p][q]  = bih[j] + bhh[j];
    }
  }
  // this lane's 8 input rows (fixed for whole kernel)
  const float* xrow = x + (size_t)(rowbase + mtile * 16 + hi * 8) * T_STEPS;
  __syncthreads();

  for (int t = 0; t < T_STEPS; ++t) {
    const __bf16* rd = hbuf[t & 1];
    __bf16*       wr = hbuf[(t + 1) & 1];
    float xr[8];
#pragma unroll
    for (int v = 0; v < 8; ++v) xr[v] = xrow[(size_t)v * T_STEPS + t];

#pragma unroll
    for (int p = 0; p < 4; ++p) {
      const int cg = cgbase + p;
      v8f acc[4];
#pragma unroll
      for (int q = 0; q < 4; ++q)
#pragma unroll
        for (int v = 0; v < 8; ++v) acc[q][v] = xr[v] * wij[p][q] + bs[p][q];

      const __bf16* hrow = rd + (mtile * 16 + l15) * 256;
#pragma unroll 1                       // keep only 4 B-frags live: no spills
      for (int kc = 0; kc < 8; ++kc) {
        const int kb = kc * 32 + hi * 8;
        Frag16 a;
        a.q[0] = *(const uint4*)(hrow + kb);
        a.q[1] = *(const uint4*)(hrow + kb + 16);
#pragma unroll
        for (int q = 0; q < 4; ++q) {
          const int j = q * 256 + cg * 16 + l15;
          const __bf16* brow = Whh + (size_t)j * 256;  // row j of W_hh == B column j
          Frag16 b;
          b.q[0] = *(const uint4*)(brow + kb);
          b.q[1] = *(const uint4*)(brow + kb + 16);
          acc[q] = __builtin_amdgcn_wmma_f32_16x16x32_bf16(
              false, a.v, false, b.v, (short)0, acc[q], false, false);
        }
      }
      // gate nonlinearities (i,f,g,o) + write new h straight to back buffer
#pragma unroll
      for (int v = 0; v < 8; ++v) {
        float ig = sigmoidf_(acc[0][v]);
        float fg = sigmoidf_(acc[1][v]);
        float gg = tanhf_(acc[2][v]);
        float og = sigmoidf_(acc[3][v]);
        float cn = fg * cst[p][v] + ig * gg;
        cst[p][v] = cn;
        wr[(mtile * 16 + v + hi * 8) * 256 + cg * 16 + l15] = f2bf(og * tanhf_(cn));
      }
    }
    __syncthreads();   // back-buffer writes complete before it becomes front
  }
  // T_STEPS even -> final h is in hbuf[0]
  for (int i = tid; i < 32 * 256; i += 256) {
    int r = i >> 8, c = i & 255;
    hout[(size_t)(rowbase + r) * 256 + c] = bf2f(hbuf[0][i]);
  }
}

// ---------------------------------------------------------------------------
// Generic bf16 GEMM: C(MxNcols,f32) = A(MxK, row-major bf16) * B with
// B supplied transposed: BT(Ncols x K, row-major). 8 waves/block, each wave
// owns one 16-row M-tile and 8 column-tile accumulators; blockIdx.y selects
// the 128-column slice (keeps worst-case live VGPRs ~150, no spills).
// ---------------------------------------------------------------------------
#define GF_RELU  1
#define GF_STF32 2
#define GF_STBF  4
#define GF_STBFT 8

__global__ void __launch_bounds__(256)
k_gemm_bf16(const __bf16* __restrict__ A, int lda,
            const __bf16* __restrict__ BT, int ldb,
            const float* __restrict__ bias,
            float* __restrict__ Cf, __bf16* __restrict__ Cb,
            int M, int Ncols, int K, int flags) {
  const int wave = threadIdx.x >> 5, lane = threadIdx.x & 31;
  const int l15 = lane & 15, hi = lane >> 4;
  const int mtile = blockIdx.x * 8 + wave;
  const int nt0 = blockIdx.y * 8;
  v8f acc[8];
  v8f zero = {};
#pragma unroll
  for (int i = 0; i < 8; ++i) acc[i] = zero;

  const __bf16* Arow = A + (size_t)(mtile * 16 + l15) * lda;
#pragma unroll 1
  for (int kc = 0; kc < K; kc += 32) {
    const int kb = kc + hi * 8;
    Frag16 a;
    a.q[0] = *(const uint4*)(Arow + kb);
    a.q[1] = *(const uint4*)(Arow + kb + 16);
    __builtin_prefetch(Arow + kb + 512, 0, 0);   // global_prefetch_b8 on stream operand
#pragma unroll
    for (int nt = 0; nt < 8; ++nt) {
      const __bf16* Brow = BT + (size_t)((nt0 + nt) * 16 + l15) * ldb;
      Frag16 b;
      b.q[0] = *(const uint4*)(Brow + kb);
      b.q[1] = *(const uint4*)(Brow + kb + 16);
      acc[nt] = __builtin_amdgcn_wmma_f32_16x16x32_bf16(
          false, a.v, false, b.v, (short)0, acc[nt], false, false);
    }
  }
#pragma unroll
  for (int nt = 0; nt < 8; ++nt) {
#pragma unroll
    for (int v = 0; v < 8; ++v) {
      const int row = mtile * 16 + v + hi * 8;
      const int col = (nt0 + nt) * 16 + l15;
      float val = acc[nt][v];
      if (bias) val += bias[col];
      if (flags & GF_RELU) val = fmaxf(val, 0.0f);
      if (flags & GF_STF32) Cf[(size_t)row * Ncols + col] = val;
      if (flags & GF_STBF)  Cb[(size_t)row * Ncols + col] = f2bf(val);
      if (flags & GF_STBFT) Cb[(size_t)col * M + row] = f2bf(val);
    }
  }
}

// ---------------------------------------------------------------------------
// Attention over 3 views + gate fusion; one thread per node, 3 passes over H2
// keeping only the 3x16 attention accumulators in registers.
// ---------------------------------------------------------------------------
__global__ void k_attn_gate(const float* __restrict__ gs, const float* __restrict__ gd,
                            const float* __restrict__ h,
                            const float* __restrict__ aW1, const float* __restrict__ ab1,
                            const float* __restrict__ aW2,
                            const float* __restrict__ gW, const float* __restrict__ gb,
                            __bf16* __restrict__ fused_bf) {
  int n = blockIdx.x * blockDim.x + threadIdx.x;
  if (n >= N_NODES) return;
  const float* gsr = gs + (size_t)n * H2C;
  const float* gdr = gd + (size_t)n * H2C;
  const float* hr  = h  + (size_t)n * H2C;

  float ta[3][ATT_HC];
#pragma unroll
  for (int a = 0; a < ATT_HC; ++a) { float b = ab1[a]; ta[0][a] = b; ta[1][a] = b; ta[2][a] = b; }
  for (int k = 0; k < H2C; ++k) {
    float s = gsr[k], d = gdr[k], c = 0.5f * (s + d);
#pragma unroll
    for (int a = 0; a < ATT_HC; ++a) {
      float w = aW1[k * ATT_HC + a];
      ta[0][a] += s * w; ta[1][a] += d * w; ta[2][a] += c * w;
    }
  }
  float wv[3];
#pragma unroll
  for (int v = 0; v < 3; ++v) {
    float acc = 0.0f;
#pragma unroll
    for (int a = 0; a < ATT_HC; ++a) acc += tanhf_(ta[v][a]) * aW2[a];
    wv[v] = acc;
  }
  float mx = fmaxf(wv[0], fmaxf(wv[1], wv[2]));
  float e0 = __expf(wv[0] - mx), e1 = __expf(wv[1] - mx), e2 = __expf(wv[2] - mx);
  float inv = rcp_(e0 + e1 + e2);
  float b0 = e0 * inv, b1 = e1 * inv, b2 = e2 * inv;

  float zacc = gb[0];
  for (int k = 0; k < H2C; ++k) {
    float s = gsr[k], d = gdr[k], c = 0.5f * (s + d);
    float g = b0 * s + b1 * d + b2 * c;
    zacc += g * gW[k] + hr[k] * gW[H2C + k];
  }
  float z = sigmoidf_(zacc);

  __bf16* fr = fused_bf + (size_t)n * H2C;
  for (int k = 0; k < H2C; ++k) {
    float s = gsr[k], d = gdr[k], c = 0.5f * (s + d);
    float g = b0 * s + b1 * d + b2 * c;
    fr[k] = f2bf(z * g + (1.0f - z) * hr[k]);
  }
}

// ---------------------------------------------------------------------------
// Sparse GCN conv: symmetric-normalized scatter with f32 atomics
// ---------------------------------------------------------------------------
__global__ void k_deg_init(float* __restrict__ deg) {
  int n = blockIdx.x * blockDim.x + threadIdx.x;
  if (n < N_NODES) deg[n] = 1.0f;   // self-loop
}
__global__ void k_deg_count(float* __restrict__ deg, const int* __restrict__ dst) {
  int e = blockIdx.x * blockDim.x + threadIdx.x;
  if (e < NEDGE) atomicAdd(&deg[dst[e]], 1.0f);
}
__global__ void k_dinv(float* __restrict__ dinv, const float* __restrict__ deg) {
  int n = blockIdx.x * blockDim.x + threadIdx.x;
  if (n < N_NODES) dinv[n] = deg[n] > 0.0f ? rsqrtf(deg[n]) : 0.0f;
}
// out[n][f] = bias[f] + xw[n][f]*dinv[n]^2   (self-loop term + init)
__global__ void k_conv_init(float* __restrict__ out, const float* __restrict__ xw,
                            const float* __restrict__ bias, const float* __restrict__ dinv,
                            int fsh) {
  int i = blockIdx.x * blockDim.x + threadIdx.x;
  int F = 1 << fsh;
  if (i < N_NODES * F) {
    int n = i >> fsh, f = i & (F - 1);
    out[i] = bias[f] + xw[i] * dinv[n] * dinv[n];
  }
}
__global__ void k_conv_scatter(float* __restrict__ out, const float* __restrict__ xw,
                               const int* __restrict__ src, const int* __restrict__ dst,
                               const float* __restrict__ dinv, int fsh) {
  int i = blockIdx.x * blockDim.x + threadIdx.x;
  int F = 1 << fsh;
  if (i < NEDGE * F) {
    int e = i >> fsh, f = i & (F - 1);
    int s = src[e], d = dst[e];
    atomicAdd(&out[(size_t)d * F + f], xw[(size_t)s * F + f] * dinv[s] * dinv[d]);
  }
}
__global__ void k_relu_bf(__bf16* __restrict__ dst, const float* __restrict__ src, int n) {
  int i = blockIdx.x * blockDim.x + threadIdx.x;
  if (i < n) dst[i] = f2bf(fmaxf(src[i], 0.0f));
}
__global__ void k_final(float* __restrict__ out, const float* __restrict__ h2,
                        const float* __restrict__ oW, const float* __restrict__ ob) {
  int n = blockIdx.x * blockDim.x + threadIdx.x;
  if (n < N_NODES) {
    float acc = ob[0];
#pragma unroll 4
    for (int f = 0; f < H4C; ++f) acc += h2[(size_t)n * H4C + f] * oW[f];
    out[n] = acc;
  }
}

// ---------------------------------------------------------------------------
// Host launcher
// ---------------------------------------------------------------------------
extern "C" void kernel_launch(void* const* d_in, const int* in_sizes, int n_in,
                              void* d_out, int out_size, void* d_ws, size_t ws_size,
                              hipStream_t stream) {
  (void)in_sizes; (void)n_in; (void)out_size; (void)ws_size;
  const float* x      = (const float*)d_in[0];
  const float* sadj   = (const float*)d_in[1];
  const float* fadj   = (const float*)d_in[2];
  const int*   ei     = (const int*)d_in[3];
  const float* W_ih   = (const float*)d_in[4];
  const float* W_hh   = (const float*)d_in[5];
  const float* b_ih   = (const float*)d_in[6];
  const float* b_hh   = (const float*)d_in[7];
  const float* sg1_W  = (const float*)d_in[8];
  const float* sg1_b  = (const float*)d_in[9];
  const float* sg2_W  = (const float*)d_in[10];
  const float* sg2_b  = (const float*)d_in[11];
  const float* dg1_W  = (const float*)d_in[12];
  const float* dg1_b  = (const float*)d_in[13];
  const float* dg2_W  = (const float*)d_in[14];
  const float* dg2_b  = (const float*)d_in[15];
  const float* att_W1 = (const float*)d_in[16];
  const float* att_b1 = (const float*)d_in[17];
  const float* att_W2 = (const float*)d_in[18];
  const float* gate_W = (const float*)d_in[19];
  const float* gate_b = (const float*)d_in[20];
  const float* g1_W   = (const float*)d_in[21];
  const float* g1_b   = (const float*)d_in[22];
  const float* g2_W   = (const float*)d_in[23];
  const float* g2_b   = (const float*)d_in[24];
  const float* out_W  = (const float*)d_in[25];
  const float* out_b  = (const float*)d_in[26];
  const int* e_src = ei;
  const int* e_dst = ei + NEDGE;

  char* ws = (char*)d_ws;
  size_t off = 0;
  auto take = [&](size_t bytes) -> void* {
    void* p = ws + off; off += (bytes + 255) & ~(size_t)255; return p;
  };
  __bf16* Whh_bf  = (__bf16*)take((size_t)1024 * 256 * 2);
  __bf16* Wsg2T   = (__bf16*)take((size_t)256 * 256 * 2);
  __bf16* Wdg2T   = (__bf16*)take((size_t)256 * 256 * 2);
  __bf16* Wg1T    = (__bf16*)take((size_t)256 * 256 * 2);
  __bf16* Wg2T    = (__bf16*)take((size_t)128 * 256 * 2);
  float*  h_lstm  = (float*)take((size_t)N_NODES * H2C * 4);
  __bf16* adjbf   = (__bf16*)take((size_t)N_NODES * N_NODES * 2);
  __bf16* t1T     = (__bf16*)take((size_t)H1C * N_NODES * 2);
  __bf16* Hbf     = (__bf16*)take((size_t)N_NODES * H1C * 2);
  __bf16* t2T     = (__bf16*)take((size_t)H2C * N_NODES * 2);
  float*  gsb     = (float*)take((size_t)N_NODES * H2C * 4);
  float*  gdb     = (float*)take((size_t)N_NODES * H2C * 4);
  __bf16* fusedbf = (__bf16*)take((size_t)N_NODES * H2C * 2);
  float*  xw1     = (float*)take((size_t)N_NODES * H3C * 4);
  float*  o1      = (float*)take((size_t)N_NODES * H3C * 4);
  __bf16* h1bf    = (__bf16*)take((size_t)N_NODES * H3C * 2);
  float*  xw2     = (float*)take((size_t)N_NODES * H4C * 4);
  float*  o2      = (float*)take((size_t)N_NODES * H4C * 4);
  float*  deg     = (float*)take((size_t)N_NODES * 4);
  float*  dinv    = (float*)take((size_t)N_NODES * 4);

  const int TB = 256;
  // --- weight conversions ---
  k_f32_to_bf16<<<(1024 * 256 / 4 + TB - 1) / TB, TB, 0, stream>>>(Whh_bf, W_hh, 1024 * 256);
  k_f32_to_bf16_T<<<(256 * 256 + TB - 1) / TB, TB, 0, stream>>>(Wsg2T, sg2_W, 256, 256);
  k_f32_to_bf16_T<<<(256 * 256 + TB - 1) / TB, TB, 0, stream>>>(Wdg2T, dg2_W, 256, 256);
  k_f32_to_bf16_T<<<(256 * 256 + TB - 1) / TB, TB, 0, stream>>>(Wg1T, g1_W, 256, 256);
  k_f32_to_bf16_T<<<(256 * 128 + TB - 1) / TB, TB, 0, stream>>>(Wg2T, g2_W, 256, 128);

  // --- LSTM over T=64 steps ---
  k_lstm<<<N_NODES / 32, 256, 0, stream>>>(x, Whh_bf, W_ih, b_ih, b_hh, h_lstm);

  const dim3 g256(N_NODES / 128, 2);   // GEMM grid for 256 output cols
  const dim3 g128(N_NODES / 128, 1);   // GEMM grid for 128 output cols

  // --- dense GCN branch: sadj ---
  k_f32_to_bf16<<<((long long)N_NODES * N_NODES / 4 + TB - 1) / TB, TB, 0, stream>>>(
      adjbf, sadj, (long long)N_NODES * N_NODES);
  k_build_t1T<<<(H1C * N_NODES + TB - 1) / TB, TB, 0, stream>>>(t1T, x, sg1_W);
  k_gemm_bf16<<<g256, 256, 0, stream>>>(adjbf, N_NODES, t1T, N_NODES, sg1_b,
      (float*)nullptr, Hbf, N_NODES, H1C, N_NODES, GF_RELU | GF_STBF);
  k_gemm_bf16<<<g256, 256, 0, stream>>>(Hbf, H1C, Wsg2T, H1C, (const float*)nullptr,
      (float*)nullptr, t2T, N_NODES, H2C, H1C, GF_STBFT);
  k_gemm_bf16<<<g256, 256, 0, stream>>>(adjbf, N_NODES, t2T, N_NODES, sg2_b,
      gsb, (__bf16*)nullptr, N_NODES, H2C, N_NODES, GF_STF32);

  // --- dense GCN branch: fadj (reuses adjbf/t1T/Hbf/t2T) ---
  k_f32_to_bf16<<<((long long)N_NODES * N_NODES / 4 + TB - 1) / TB, TB, 0, stream>>>(
      adjbf, fadj, (long long)N_NODES * N_NODES);
  k_build_t1T<<<(H1C * N_NODES + TB - 1) / TB, TB, 0, stream>>>(t1T, x, dg1_W);
  k_gemm_bf16<<<g256, 256, 0, stream>>>(adjbf, N_NODES, t1T, N_NODES, dg1_b,
      (float*)nullptr, Hbf, N_NODES, H1C, N_NODES, GF_RELU | GF_STBF);
  k_gemm_bf16<<<g256, 256, 0, stream>>>(Hbf, H1C, Wdg2T, H1C, (const float*)nullptr,
      (float*)nullptr, t2T, N_NODES, H2C, H1C, GF_STBFT);
  k_gemm_bf16<<<g256, 256, 0, stream>>>(adjbf, N_NODES, t2T, N_NODES, dg2_b,
      gdb, (__bf16*)nullptr, N_NODES, H2C, N_NODES, GF_STF32);

  // --- attention + gate fusion ---
  k_attn_gate<<<(N_NODES + TB - 1) / TB, TB, 0, stream>>>(
      gsb, gdb, h_lstm, att_W1, att_b1, att_W2, gate_W, gate_b, fusedbf);

  // --- sparse GCN layer 1 ---
  k_gemm_bf16<<<g256, 256, 0, stream>>>(fusedbf, H2C, Wg1T, H2C,
      (const float*)nullptr, xw1, (__bf16*)nullptr, N_NODES, H3C, H2C, GF_STF32);
  k_deg_init<<<(N_NODES + TB - 1) / TB, TB, 0, stream>>>(deg);
  k_deg_count<<<(NEDGE + TB - 1) / TB, TB, 0, stream>>>(deg, e_dst);
  k_dinv<<<(N_NODES + TB - 1) / TB, TB, 0, stream>>>(dinv, deg);
  k_conv_init<<<(N_NODES * H3C + TB - 1) / TB, TB, 0, stream>>>(o1, xw1, g1_b, dinv, 8);
  k_conv_scatter<<<(NEDGE * H3C + TB - 1) / TB, TB, 0, stream>>>(o1, xw1, e_src, e_dst, dinv, 8);
  k_relu_bf<<<(N_NODES * H3C + TB - 1) / TB, TB, 0, stream>>>(h1bf, o1, N_NODES * H3C);

  // --- sparse GCN layer 2 ---
  k_gemm_bf16<<<g128, 256, 0, stream>>>(h1bf, H3C, Wg2T, H3C,
      (const float*)nullptr, xw2, (__bf16*)nullptr, N_NODES, H4C, H3C, GF_STF32);
  k_conv_init<<<(N_NODES * H4C + TB - 1) / TB, TB, 0, stream>>>(o2, xw2, g2_b, dinv, 7);
  k_conv_scatter<<<(NEDGE * H4C + TB - 1) / TB, TB, 0, stream>>>(o2, xw2, e_src, e_dst, dinv, 7);

  // --- output head ---
  k_final<<<(N_NODES + TB - 1) / TB, TB, 0, stream>>>((float*)d_out, o2, out_W, out_b);
}
